// lossy_Conv2d_new_64707977282056
// MI455X (gfx1250) — compile-verified, matrix operand order checked by reading, not run on hardware
//
#include <hip/hip_runtime.h>
#include <hip/hip_bf16.h>
#include <stdint.h>

// CDNA5 (gfx1250) wave32 WMMA types
typedef __attribute__((ext_vector_type(16))) __bf16       v16bf;
typedef __attribute__((ext_vector_type(8)))  float        v8f;
typedef __attribute__((ext_vector_type(8)))  unsigned int v8u;
// TDM descriptor vector types (per CDNA5_HIP.md probe)
typedef __attribute__((ext_vector_type(4)))  unsigned int u32x4;
typedef __attribute__((ext_vector_type(8)))  int          i32x8;
typedef __attribute__((ext_vector_type(4)))  int          i32x4;

#if defined(__has_builtin)
#if __has_builtin(__builtin_amdgcn_tensor_load_to_lds)
#define HAVE_TDM 1
#endif
#endif
#ifndef HAVE_TDM
#define HAVE_TDM 0
#endif

// float -> bf16 bits, round-to-nearest-even (scalar fallback / weight prep)
__device__ __forceinline__ unsigned short f2bf(float f) {
  unsigned u = __builtin_bit_cast(unsigned, f);
  unsigned r = u + 0x7FFFu + ((u >> 16) & 1u);
  return (unsigned short)(r >> 16);
}

// two floats -> packed bf16x2 dword (hardware packed cvt where available)
__device__ __forceinline__ unsigned pk_bf16(float lo, float hi) {
  __hip_bfloat162 h2 = __float22bfloat162_rn(float2{lo, hi});
  unsigned u;
  __builtin_memcpy(&u, &h2, sizeof(u));  // __hip_bfloat162 not trivially copyable
  return u;
}

// deterministic per-pixel hash for the Bernoulli(0.5) halo mask
__device__ __forceinline__ unsigned pcg_hash(unsigned v) {
  unsigned s = v * 747796405u + 2891336453u;
  unsigned w = ((s >> ((s >> 28u) + 4u)) ^ s) * 277803737u;
  return (w >> 22u) ^ w;
}

// Stage 0: repack weights (64,64,3,3) f32 -> bf16 [m=cout][k = (ky*3+kx)*64 + ci]
__global__ void prep_weights_kernel(const float* __restrict__ w,
                                    unsigned short* __restrict__ wp) {
  int i = blockIdx.x * blockDim.x + threadIdx.x;
  if (i >= 64 * 576) return;
  int m  = i / 576, k = i - m * 576;
  int s  = k >> 6,  ci = k & 63;       // s = ky*3+kx
  int ky = s / 3,   kx = s - ky * 3;
  wp[i] = f2bf(w[((m * 64 + ci) * 3 + ky) * 3 + kx]);
}

#if HAVE_TDM
// Issue one TDM 2D tile load: 32(k) x 64(m) bf16 slice of wp, stride 576,
// into LDS rows of 20 dwords (16 data + 4 pad): pad_interval=16dw, pad=4dw.
__device__ __forceinline__ void tdm_weight_slice(const unsigned short* wp,
                                                 int kk, void* lds) {
  unsigned lds_addr = (unsigned)(size_t)lds;
  unsigned long long ga = (unsigned long long)(size_t)(wp + kk);
  u32x4 g0;
  g0[0] = 1u;                                   // count=1 (valid)
  g0[1] = lds_addr;                             // lds_addr (bytes)
  g0[2] = (unsigned)ga;                         // global_addr[31:0]
  g0[3] = (unsigned)((ga >> 32) & 0x01FFFFFFu)  // global_addr[56:32]
          | (2u << 30);                         // type=2 (image)
  i32x8 g1;
  g1[0] = (int)0x06D10000u;  // data_size=2B | pad_en | pad_int=16dw | pad=4dw
  g1[1] = (int)(576u << 16); // tensor_dim0[15:0]=576
  g1[2] = (int)(64u << 16);  // tensor_dim0 hi | tensor_dim1[15:0]=64
  g1[3] = (int)(32u << 16);  // tensor_dim1 hi | tile_dim0=32
  g1[4] = 64;                // tile_dim1=64, tile_dim2=0
  g1[5] = 576;               // tensor_dim0_stride lo
  g1[6] = 0;
  g1[7] = 0;
  i32x4 g2 = {0, 0, 0, 0};
  i32x4 g3 = {0, 0, 0, 0};
#if __clang_major__ >= 23
  i32x8 g4 = {0, 0, 0, 0, 0, 0, 0, 0};
  __builtin_amdgcn_tensor_load_to_lds(g0, g1, g2, g3, g4, 0);
#else
  __builtin_amdgcn_tensor_load_to_lds(g0, g1, g2, g3, 0);
#endif
}
#endif

__device__ __forceinline__ v8u lds_frag(const uint4* base, int row, int kh) {
  uint4 a = base[row * 5 + kh];      // dwords kh*4 + 0..3  (k pairs 0..7)
  uint4 b = base[row * 5 + kh + 2];  // dwords kh*4 + 8..11 (k pairs 16..23)
  v8u r;
  r[0] = a.x; r[1] = a.y; r[2] = a.z; r[3] = a.w;
  r[4] = b.x; r[5] = b.y; r[6] = b.z; r[7] = b.w;
  return r;
}

#define WMMA_BF16(A, B, C) \
  __builtin_amdgcn_wmma_f32_16x16x32_bf16(false, (A), false, (B), (short)0, (C), false, false)

// Stage 1: implicit-GEMM lossy conv.
// Block: 128 threads (4 waves) -> 64 cout x 64 pixels of one (batch, quadrant).
// GEMM: M=64, N=64, K=576 (tap-major). Wave w owns a 2x2 grid of 16x16 tiles
// -> 4 WMMAs per K=32 step, 18 steps. Double-buffered LDS, one barrier/step;
// TDM prefetches the next weight slice, overlapping DMA with WMMA + im2col.
// #pragma unroll 2 keeps the double-buffer parity static without blowing I$.
__global__ __launch_bounds__(128) void lossy_conv_wmma_kernel(
    const float* __restrict__ x, const unsigned short* __restrict__ wp,
    const float* __restrict__ bias, float* __restrict__ out) {
  // rows: 5 uint4 = 20 dwords (16 data + 4 pad) -> b128-aligned, conflict-free
  __shared__ uint4 Wl[2][64 * 5];  // weight slice  [m][k]
  __shared__ uint4 Pl[2][64 * 5];  // im2col patch  [n][k]

  const int tid   = threadIdx.x;
  const int lane  = tid & 31;
  const int wv    = tid >> 5;          // 0..3
  const int mBase = (wv & 1) << 5;     // 0 or 32 (two 16-row tiles)
  const int nBase = (wv >> 1) << 5;    // 0 or 32 (two 16-col tiles)

  int bid = blockIdx.x;
  int pb  = bid % 49;                  // 56*56/64 pixel strips per quadrant
  int t   = bid / 49;
  int q   = t & 3;
  int b   = t >> 2;
  const int p0  = pb << 6;
  const int qy0 = (q >> 1) * 56;
  const int qx0 = (q & 1) * 56;

  // im2col role: thread -> (pixel n, 16 consecutive ci)
  const int pn  = tid & 63;            // 0..63
  const int jb2 = tid >> 6;            // 0..1 -> ci sub-chunk of 16
  const int pix = p0 + pn;
  const int oy  = pix / 56;
  const int ox  = pix - oy * 56;
  const int gy0 = qy0 + oy - 1;
  const int gx0 = qx0 + ox - 1;

  // fragment addressing (loop-invariant)
  const int ml = mBase + (lane & 15);
  const int nl = nBase + (lane & 15);
  const int kh = lane >> 4;            // k-half select (0 or 1)

#if !HAVE_TDM
  const int wm  = tid >> 1;            // 0..63
  const int wko = (tid & 1) << 4;      // 0 or 16 bf16 elements
#endif

  v8f acc00 = {}, acc01 = {}, acc10 = {}, acc11 = {};

#if HAVE_TDM
  if (wv == 0) tdm_weight_slice(wp, 0, &Wl[0][0]);
#endif

#pragma unroll 2
  for (int it = 0; it < 18; ++it) {
    const int kk     = it << 5;
    const int buf    = it & 1;
    const int s      = kk >> 6;        // tap = ky*3+kx
    const int ciBase = kk & 63;        // 0 or 32
    const int ky     = s / 3;
    const int kx     = s - ky * 3;

#if !HAVE_TDM
    {  // cooperative weight slice copy (2 x b128 per thread)
      const uint4* g = (const uint4*)(wp + wm * 576 + kk + wko);
      Wl[buf][wm * 5 + (wko >> 3)]     = g[0];
      Wl[buf][wm * 5 + (wko >> 3) + 1] = g[1];
      if (kk + 32 < 576)
        __builtin_prefetch(wp + wm * 576 + kk + 32 + wko, 0, 1);
    }
#endif

    // ---- im2col: fixed tap; 16 channels per thread; hoisted bounds ----
    const int gy = gy0 + ky;
    const int gx = gx0 + kx;
    float vals[16];
    if (gy >= 0 && gy < 112 && gx >= 0 && gx < 112) {
      const int ci0  = ciBase + (jb2 << 4);
      const int idx0 = ((b * 64 + ci0) * 112 + gy) * 112 + gx;
#pragma unroll
      for (int j = 0; j < 16; ++j) vals[j] = x[idx0 + j * 12544];
      const bool inQ = (gy >= qy0) && (gy < qy0 + 56) &&
                       (gx >= qx0) && (gx < qx0 + 56);
      if (!inQ) {  // halo ring: Bernoulli(0.5) keep-mask per (q,b,ci,gy,gx)
        const unsigned sb = (unsigned)((q * 32 + b) * 64 + ci0) * 12544u +
                            (unsigned)(gy * 112 + gx);
#pragma unroll
        for (int j = 0; j < 16; ++j)
          if (!(pcg_hash(sb + (unsigned)j * 12544u) & 1u)) vals[j] = 0.f;
      }
    } else {
#pragma unroll
      for (int j = 0; j < 16; ++j) vals[j] = 0.f;
    }
    uint4 d0, d1;
    d0.x = pk_bf16(vals[0],  vals[1]);
    d0.y = pk_bf16(vals[2],  vals[3]);
    d0.z = pk_bf16(vals[4],  vals[5]);
    d0.w = pk_bf16(vals[6],  vals[7]);
    d1.x = pk_bf16(vals[8],  vals[9]);
    d1.y = pk_bf16(vals[10], vals[11]);
    d1.z = pk_bf16(vals[12], vals[13]);
    d1.w = pk_bf16(vals[14], vals[15]);
    Pl[buf][pn * 5 + (jb2 << 1)]     = d0;
    Pl[buf][pn * 5 + (jb2 << 1) + 1] = d1;

#if HAVE_TDM
    if (wv == 0) __builtin_amdgcn_s_wait_tensorcnt(0);  // slice `it` landed
#endif
    __syncthreads();  // single barrier per step (double-buffered LDS)

#if HAVE_TDM
    // prefetch next weight slice into the idle buffer; overlaps with compute
    if (wv == 0 && it + 1 < 18)
      tdm_weight_slice(wp, (it + 1) << 5, &Wl[1 - buf][0]);
#endif

    // ---- fragments: 2 x ds_load_b128 each; 4 WMMAs per step ----
    v16bf A0 = __builtin_bit_cast(v16bf, lds_frag(&Wl[buf][0], ml,      kh));
    v16bf A1 = __builtin_bit_cast(v16bf, lds_frag(&Wl[buf][0], ml + 16, kh));
    v16bf B0 = __builtin_bit_cast(v16bf, lds_frag(&Pl[buf][0], nl,      kh));
    v16bf B1 = __builtin_bit_cast(v16bf, lds_frag(&Pl[buf][0], nl + 16, kh));
    acc00 = WMMA_BF16(A0, B0, acc00);
    acc01 = WMMA_BF16(A0, B1, acc01);
    acc10 = WMMA_BF16(A1, B0, acc10);
    acc11 = WMMA_BF16(A1, B1, acc11);
  }

  // ---- epilogue: four 16x16 tiles; D: lanes 0-15 M=v, lanes 16-31 M=v+8 ----
  const int mAdd = (lane >> 4) << 3;
#pragma unroll
  for (int mt = 0; mt < 2; ++mt) {
#pragma unroll
    for (int nt = 0; nt < 2; ++nt) {
      const v8f* a = (mt == 0) ? (nt == 0 ? &acc00 : &acc01)
                               : (nt == 0 ? &acc10 : &acc11);
      const int nOut = nBase + nt * 16 + (lane & 15);
      const int pixO = p0 + nOut;
      const int oyO  = pixO / 56;
      const int oxO  = pixO - oyO * 56;
      const int gy   = qy0 + oyO;
      const int gx   = qx0 + oxO;
#pragma unroll
      for (int v = 0; v < 8; ++v) {
        int co = mBase + mt * 16 + v + mAdd;
        out[((b * 64 + co) * 112 + gy) * 112 + gx] = (*a)[v] + bias[co];
      }
    }
  }
}

extern "C" void kernel_launch(void* const* d_in, const int* in_sizes, int n_in,
                              void* d_out, int out_size, void* d_ws,
                              size_t ws_size, hipStream_t stream) {
  (void)in_sizes; (void)n_in; (void)out_size; (void)ws_size;
  const float* x  = (const float*)d_in[0];     // (32,64,112,112)
  const float* w  = (const float*)d_in[1];     // (64,64,3,3)
  const float* bs = (const float*)d_in[2];     // (64,)
  float* out = (float*)d_out;                  // (32,64,112,112)
  unsigned short* wp = (unsigned short*)d_ws;  // 64*576 bf16 = 72 KB

  prep_weights_kernel<<<(64 * 576 + 255) / 256, 256, 0, stream>>>(w, wp);
  lossy_conv_wmma_kernel<<<32 * 4 * 49, 128, 0, stream>>>(x, wp, bs, out);
}
